// MultiHeadSelfAttention_72395968741921
// MI455X (gfx1250) — matrix operand order repaired and checked by reading, not compiled, not run on hardware
//
#include <hip/hip_runtime.h>

#define DEV __device__ __forceinline__

typedef __bf16 bf16_t;
typedef __attribute__((ext_vector_type(16))) __bf16 v16bf;
typedef __attribute__((ext_vector_type(8)))  float  v8f;
typedef __attribute__((ext_vector_type(4)))  int    v4i;

constexpr int Bc = 4, Sc = 2048, Dc = 1024, Hc = 16, DKc = 64;
constexpr int M_TOTAL = Bc * Sc;          // 8192

union Frag16 { v16bf v; v4i i[2]; };

DEV v8f zero8() {
  v8f z;
#pragma unroll
  for (int i = 0; i < 8; ++i) z[i] = 0.f;
  return z;
}

// A-matrix fragment (16x32 bf16), per ISA 7.12.2:
// lane L: row M=L&15; halfs K=[k0+sel*8 .. +8) and K=[k0+16+sel*8 .. +8), sel=L>>4
DEV v16bf load_a_frag(const bf16_t* __restrict__ base, int ld, int row0, int k0, int lane) {
  const int r   = row0 + (lane & 15);
  const int sel = (lane >> 4) & 1;
  const bf16_t* p = base + (size_t)r * ld + k0 + sel * 8;
  Frag16 u;
  u.i[0] = *(const v4i*)(p);
  u.i[1] = *(const v4i*)(p + 16);
  return u.v;
}

// B-matrix fragment (32x16 bf16): lane L holds column N=L&15,
// K=[k0+sel*8 .. +8) and K=[k0+16+sel*8 .. +8) (column c of B == row c of row-major [N x K]).
DEV v16bf load_b_frag(const bf16_t* __restrict__ base, int ld, int col0, int k0, int lane) {
  const int c   = col0 + (lane & 15);
  const int sel = (lane >> 4) & 1;
  const bf16_t* p = base + (size_t)c * ld + k0 + sel * 8;
  Frag16 u;
  u.i[0] = *(const v4i*)(p);
  u.i[1] = *(const v4i*)(p + 16);
  return u.v;
}

DEV v8f wmma_bf16(v16bf a, v16bf b, v8f c) {
  return __builtin_amdgcn_wmma_f32_16x16x32_bf16(false, a, false, b, (short)0, c, false, false);
}

// CDNA5 async copy: global -> LDS, tracked by ASYNCcnt, no VGPR staging.
DEV void async_copy_b128(unsigned lds_byte_addr, const bf16_t* g) {
  asm volatile("global_load_async_to_lds_b128 %0, %1, off"
               :: "v"(lds_byte_addr), "v"((unsigned long long)(size_t)g)
               : "memory");
}
DEV void wait_async_le8() { asm volatile("s_wait_asynccnt 0x8" ::: "memory"); }
DEV void wait_async_0()   { asm volatile("s_wait_asynccnt 0x0" ::: "memory"); }
DEV void wait_ds_0()      { asm volatile("s_wait_dscnt 0x0"    ::: "memory"); }

// ---------------- fp32 -> bf16 conversion ----------------
__global__ __launch_bounds__(256)
void cvt_f32_bf16_kernel(const float* __restrict__ in, bf16_t* __restrict__ out, int n) {
  int i = blockIdx.x * 256 + threadIdx.x;
  if (i < n) out[i] = (bf16_t)in[i];
}

// ---------------- async double-buffered WMMA GEMM: Y[M,N] = A[M,K] * W[N,K]^T --------
// Block tile 128x128, K-chunk 64, LDS double buffer staged with async-to-LDS copies.
// MODE 0: bf16 out (B,H,S,Dk) | MODE 1: bf16 out (B,H,Dk,S) | MODE 2: fp32 row-major
template <int MODE>
__global__ __launch_bounds__(256)
void gemm_wmma_kernel(const bf16_t* __restrict__ A, const bf16_t* __restrict__ W,
                      void* __restrict__ out, int M, int N, int K) {
  __shared__ __align__(16) bf16_t As[2][128][64];   // 2 x 16KB
  __shared__ __align__(16) bf16_t Bs[2][128][64];   // 2 x 16KB

  const int t    = threadIdx.x;
  const int lane = t & 31;
  const int wave = t >> 5;                          // 8 waves: 4 (rows) x 2 (cols)
  const int row_blk = blockIdx.x * 128;
  const int col_blk = blockIdx.y * 128;
  const int wr = (wave >> 1) * 32;                  // wave row offset in tile
  const int wc = (wave & 1) * 64;                   // wave col offset in tile

  // Each thread copies 4 x 16B of A and 4 x 16B of B per stage (8 async ops/wave).
  auto stage = [&](int buf, int k0) {
#pragma unroll
    for (int q = 0; q < 4; ++q) {
      int s   = t + q * 256;        // 0..1023
      int r   = s >> 3;             // 0..127
      int seg = (s & 7) * 8;        // half offset within 64-half row
      async_copy_b128((unsigned)(size_t)&As[buf][r][seg],
                      A + (size_t)(row_blk + r) * K + k0 + seg);
      async_copy_b128((unsigned)(size_t)&Bs[buf][r][seg],
                      W + (size_t)(col_blk + r) * K + k0 + seg);
    }
  };

  v8f acc[2][4];
#pragma unroll
  for (int r = 0; r < 2; ++r)
#pragma unroll
    for (int j = 0; j < 4; ++j) acc[r][j] = zero8();

  const int nkc = K / 64;
  stage(0, 0);
  for (int kc = 0; kc < nkc; ++kc) {
    const int buf = kc & 1;
    if (kc + 1 < nkc) {
      stage(buf ^ 1, (kc + 1) * 64);   // prefetch next chunk (8 more async ops)
      wait_async_le8();                // current chunk's 8 ops done (in-order)
    } else {
      wait_async_0();
    }
    __syncthreads();

#pragma unroll
    for (int ks = 0; ks < 2; ++ks) {
      v16bf a0 = load_a_frag(&As[buf][0][0], 64, wr,      ks * 32, lane);
      v16bf a1 = load_a_frag(&As[buf][0][0], 64, wr + 16, ks * 32, lane);
      v16bf b0 = load_b_frag(&Bs[buf][0][0], 64, wc +  0, ks * 32, lane);
      v16bf b1 = load_b_frag(&Bs[buf][0][0], 64, wc + 16, ks * 32, lane);
      v16bf b2 = load_b_frag(&Bs[buf][0][0], 64, wc + 32, ks * 32, lane);
      v16bf b3 = load_b_frag(&Bs[buf][0][0], 64, wc + 48, ks * 32, lane);
      acc[0][0] = wmma_bf16(a0, b0, acc[0][0]);
      acc[1][0] = wmma_bf16(a1, b0, acc[1][0]);
      acc[0][1] = wmma_bf16(a0, b1, acc[0][1]);
      acc[1][1] = wmma_bf16(a1, b1, acc[1][1]);
      acc[0][2] = wmma_bf16(a0, b2, acc[0][2]);
      acc[1][2] = wmma_bf16(a1, b2, acc[1][2]);
      acc[0][3] = wmma_bf16(a0, b3, acc[0][3]);
      acc[1][3] = wmma_bf16(a1, b3, acc[1][3]);
    }
    __syncthreads();
  }

  // C layout: vgpr i, lane L -> m = 8*(L>>4)+i, n = L&15
#pragma unroll
  for (int r = 0; r < 2; ++r)
#pragma unroll
    for (int j = 0; j < 4; ++j)
#pragma unroll
      for (int i = 0; i < 8; ++i) {
        int m = row_blk + wr + 16 * r + 8 * (lane >> 4) + i;
        int n = col_blk + wc + 16 * j + (lane & 15);
        float v = acc[r][j][i];
        if (MODE == 0) {
          int b = m / Sc, s = m % Sc, h = n >> 6, d = n & 63;
          ((bf16_t*)out)[((((size_t)b * Hc + h) * Sc) + s) * DKc + d] = (bf16_t)v;
        } else if (MODE == 1) {
          int b = m / Sc, s = m % Sc, h = n >> 6, d = n & 63;
          ((bf16_t*)out)[((((size_t)b * Hc + h) * DKc) + d) * Sc + s] = (bf16_t)v;
        } else {
          ((float*)out)[(size_t)m * N + n] = v;
        }
      }
}

// ---------------- flash attention (causal, online softmax) ----------------
// grid = (S/64, H, B), block = 128 (4 waves); each wave owns 16 query rows.
// K/V tiles (64x64) are shared by all 4 waves -> staged in LDS with async
// copies, double-buffered across the key-tile loop.
// Q,K in (B,H,S,Dk) bf16; V in (B,H,Dk,S) bf16; out bf16 (B,S,H,Dk) = (B,S,D).
__global__ __launch_bounds__(128)
void flash_attn_kernel(const bf16_t* __restrict__ Q, const bf16_t* __restrict__ Km,
                       const bf16_t* __restrict__ Vt, bf16_t* __restrict__ O) {
  __shared__ __align__(16) bf16_t Ks[2][64][64];    // 2 x 8KB  (rows = key, cols = Dk)
  __shared__ __align__(16) bf16_t Vs[2][64][64];    // 2 x 8KB  (rows = d,   cols = key)
  __shared__ __align__(16) float  sS[4][16][64];
  __shared__ __align__(16) bf16_t sP[4][16][64];
  __shared__ float sCorr[4][16];
  __shared__ float sLinv[4][16];

  const int t    = threadIdx.x;
  const int lane = t & 31;
  const int wave = t >> 5;
  const int qt = blockIdx.x;   // query tile (64 rows)
  const int h  = blockIdx.y;
  const int b  = blockIdx.z;

  const bf16_t* Qh = Q  + (((size_t)b * Hc + h) * Sc) * DKc;
  const bf16_t* Kh = Km + (((size_t)b * Hc + h) * Sc) * DKc;
  const bf16_t* Vh = Vt + (((size_t)b * Hc + h) * DKc) * Sc;

  const int q0 = qt * 64 + wave * 16;  // first query row owned by this wave

  // Stage one 64x64 K tile + 64x64 V^T tile: 128 threads x (4+4) b128 async ops.
  auto stageKV = [&](int buf, int kt2) {
    const int kb = kt2 * 64;
#pragma unroll
    for (int q = 0; q < 4; ++q) {
      int s   = t + q * 128;          // 0..511
      int r   = s >> 3;               // 0..63
      int seg = (s & 7) * 8;          // half offset within 64-half row
      async_copy_b128((unsigned)(size_t)&Ks[buf][r][seg],
                      Kh + (size_t)(kb + r) * DKc + seg);
      async_copy_b128((unsigned)(size_t)&Vs[buf][r][seg],
                      Vh + (size_t)r * Sc + kb + seg);
    }
  };

  v16bf qf[2];
  qf[0] = load_a_frag(Qh, DKc, q0, 0, lane);
  qf[1] = load_a_frag(Qh, DKc, q0, 32, lane);

  v8f o[4];
#pragma unroll
  for (int j = 0; j < 4; ++j) o[j] = zero8();

  float m_run = -1e30f, l_run = 0.f;   // valid in lanes 0..15 (row = lane)
  const float scale = 0.125f;          // 1/sqrt(64)

  stageKV(0, 0);
  for (int kt = 0; kt <= qt; ++kt) {
    const int buf = kt & 1;
    if (kt + 1 <= qt) {
      stageKV(buf ^ 1, kt + 1);        // prefetch next K/V tile
      wait_async_le8();
    } else {
      wait_async_0();
    }
    __syncthreads();

    const int kbase = kt * 64;
    v8f sacc[4];
#pragma unroll
    for (int j = 0; j < 4; ++j) sacc[j] = zero8();

    // scores = Q * K^T from LDS tile
#pragma unroll
    for (int kc = 0; kc < 2; ++kc)
#pragma unroll
      for (int j = 0; j < 4; ++j) {
        v16bf kf = load_b_frag(&Ks[buf][0][0], 64, 16 * j, kc * 32, lane);
        sacc[j] = wmma_bf16(qf[kc], kf, sacc[j]);
      }

    // scale + causal mask + stage scores to LDS (fp32)
#pragma unroll
    for (int j = 0; j < 4; ++j)
#pragma unroll
      for (int i = 0; i < 8; ++i) {
        int mloc = 8 * (lane >> 4) + i;
        int nloc = 16 * j + (lane & 15);
        float v = sacc[j][i] * scale;
        if (kbase + nloc > q0 + mloc) v = -1e30f;  // only bites on diagonal tile
        sS[wave][mloc][nloc] = v;
      }
    wait_ds_0();

    // online softmax: lanes 0..15 each own one row
    if (lane < 16) {
      const int r = lane;
      float mloc = -1e30f;
      for (int n = 0; n < 64; ++n) mloc = fmaxf(mloc, sS[wave][r][n]);
      float mnew = fmaxf(m_run, mloc);
      float corr = __expf(m_run - mnew);
      float ssum = 0.f;
      for (int n = 0; n < 64; ++n) {
        float p = __expf(sS[wave][r][n] - mnew);
        ssum += p;
        sP[wave][r][n] = (bf16_t)p;
      }
      l_run = l_run * corr + ssum;
      m_run = mnew;
      sCorr[wave][r] = corr;
    }
    wait_ds_0();

    // rescale running output accumulators
#pragma unroll
    for (int i = 0; i < 8; ++i) {
      float f = sCorr[wave][8 * (lane >> 4) + i];
#pragma unroll
      for (int j = 0; j < 4; ++j) o[j][i] *= f;
    }

    // O += P * V from LDS tiles (P as A-frag, V^T rows as B-frag)
#pragma unroll
    for (int kc = 0; kc < 2; ++kc) {
      v16bf pf = load_a_frag(&sP[wave][0][0], 64, 0, kc * 32, lane);
#pragma unroll
      for (int j = 0; j < 4; ++j) {
        v16bf vf = load_b_frag(&Vs[buf][0][0], 64, 16 * j, kc * 32, lane);
        o[j] = wmma_bf16(pf, vf, o[j]);
      }
    }
    __syncthreads();   // tile consumed; safe to overwrite next iteration
  }

  if (lane < 16) sLinv[wave][lane] = 1.f / l_run;
  wait_ds_0();

  // write (B,S,H,Dk) bf16
#pragma unroll
  for (int i = 0; i < 8; ++i) {
    int mloc = 8 * (lane >> 4) + i;
    float linv = sLinv[wave][mloc];
    int s = q0 + mloc;
#pragma unroll
    for (int j = 0; j < 4; ++j) {
      int d = 16 * j + (lane & 15);
      O[((((size_t)b * Sc + s) * Hc) + h) * DKc + d] = (bf16_t)(o[j][i] * linv);
    }
  }
}

// ---------------- host-side orchestration ----------------
extern "C" void kernel_launch(void* const* d_in, const int* in_sizes, int n_in,
                              void* d_out, int out_size, void* d_ws, size_t ws_size,
                              hipStream_t stream) {
  (void)in_sizes; (void)n_in; (void)out_size; (void)ws_size;
  const float* x  = (const float*)d_in[0];
  const float* Wq = (const float*)d_in[1];
  const float* Wk = (const float*)d_in[2];
  const float* Wv = (const float*)d_in[3];
  const float* Wo = (const float*)d_in[4];
  float* out = (float*)d_out;

  char* ws = (char*)d_ws;
  size_t off = 0;
  bf16_t* xb  = (bf16_t*)(ws + off); off += (size_t)M_TOTAL * Dc * 2;   // reused as attn-out
  bf16_t* wqb = (bf16_t*)(ws + off); off += (size_t)Dc * Dc * 2;
  bf16_t* wkb = (bf16_t*)(ws + off); off += (size_t)Dc * Dc * 2;
  bf16_t* wvb = (bf16_t*)(ws + off); off += (size_t)Dc * Dc * 2;
  bf16_t* wob = (bf16_t*)(ws + off); off += (size_t)Dc * Dc * 2;
  bf16_t* Qb  = (bf16_t*)(ws + off); off += (size_t)M_TOTAL * Dc * 2;
  bf16_t* Kb  = (bf16_t*)(ws + off); off += (size_t)M_TOTAL * Dc * 2;
  bf16_t* Vtb = (bf16_t*)(ws + off); off += (size_t)M_TOTAL * Dc * 2;

  const int nX = M_TOTAL * Dc;       // 8,388,608
  const int nW = Dc * Dc;            // 1,048,576
  cvt_f32_bf16_kernel<<<(nX + 255) / 256, 256, 0, stream>>>(x,  xb,  nX);
  cvt_f32_bf16_kernel<<<(nW + 255) / 256, 256, 0, stream>>>(Wq, wqb, nW);
  cvt_f32_bf16_kernel<<<(nW + 255) / 256, 256, 0, stream>>>(Wk, wkb, nW);
  cvt_f32_bf16_kernel<<<(nW + 255) / 256, 256, 0, stream>>>(Wv, wvb, nW);
  cvt_f32_bf16_kernel<<<(nW + 255) / 256, 256, 0, stream>>>(Wo, wob, nW);

  dim3 ggrid(M_TOTAL / 128, Dc / 128);   // 64 x 8
  gemm_wmma_kernel<0><<<ggrid, 256, 0, stream>>>(xb, wqb, Qb,  M_TOTAL, Dc, Dc);
  gemm_wmma_kernel<0><<<ggrid, 256, 0, stream>>>(xb, wkb, Kb,  M_TOTAL, Dc, Dc);
  gemm_wmma_kernel<1><<<ggrid, 256, 0, stream>>>(xb, wvb, Vtb, M_TOTAL, Dc, Dc);

  bf16_t* Ob = xb;  // reuse: x_bf16 no longer needed after projections
  flash_attn_kernel<<<dim3(Sc / 64, Hc, Bc), 128, 0, stream>>>(Qb, Kb, Vtb, Ob);

  gemm_wmma_kernel<2><<<ggrid, 256, 0, stream>>>(Ob, wob, (void*)out, M_TOTAL, Dc, Dc);
}
